// LinearQuantizerModel_33191507264088
// MI455X (gfx1250) — matrix operand order repaired and checked by reading, not compiled, not run on hardware
//
#include <hip/hip_runtime.h>
#include <hip/hip_bf16.h>

// ---------------------------------------------------------------------------
// CDNA5 (gfx1250) implementation: 3-layer MLP (f16 WMMA, f32 accum) + argmax
// + bounded cyclic forward-fill.  Wave32, V_WMMA_F32_16X16X32_F16 path.
// v2: LDS double-buffered B tiles (kills 8x redundant B traffic and the
//     s_wait_loadcnt 0 serialization), 32x64 per-wave output tiles.
// ---------------------------------------------------------------------------

typedef __attribute__((ext_vector_type(16))) _Float16 v16h;
typedef __attribute__((ext_vector_type(8)))  _Float16 v8h;
typedef __attribute__((ext_vector_type(8)))  float    v8f;

#define STOP_TOK 500
#define FILL_ITERS 10000

static const int M_ROWS = 16 * 2000;              // 32000 tokens
static const int K1 = 1024, N1 = 894, N1P = 896;  // layer 1 (pad N to 64x)
static const int K2 = 896,  N2 = 763, N2P = 768;  // layer 2 (K = padded N1)
static const int K3 = 768,  N3 = 501, N3P = 512;  // layer 3

// workspace layout (bytes, 256-aligned)
static const size_t OFF_XH   = 0;                                   // 32000*1024*2
static const size_t OFF_W1T  = OFF_XH   + (size_t)M_ROWS * K1 * 2;  // 896*1024*2
static const size_t OFF_W2T  = OFF_W1T  + (size_t)N1P * K1 * 2;     // 768*896*2
static const size_t OFF_W3T  = OFF_W2T  + (size_t)N2P * K2 * 2;     // 512*768*2
static const size_t OFF_H1   = OFF_W3T  + (size_t)N3P * K3 * 2;     // 32000*896*2
static const size_t OFF_H2   = OFF_H1   + (size_t)M_ROWS * N1P * 2; // 32000*768*2
static const size_t OFF_CODE = OFF_H2   + (size_t)M_ROWS * N2P * 2; // 32000*4

union AFrag { v16h v; v8h h[2]; };

__device__ inline v8f wmma_f16(v16h a, v16h b, v8f c) {
    // D = A(16x32 f16) * B(32x16 f16) + C(16x16 f32)
    return __builtin_amdgcn_wmma_f32_16x16x32_f16(
        /*neg_a=*/false, a, /*neg_b=*/false, b,
        /*c_mod=*/(short)0, c, /*reuse_a=*/false, /*reuse_b=*/false);
}

// -------------------------- prep kernels -----------------------------------

__global__ void cvt_f32_to_f16(const float* __restrict__ src,
                               _Float16* __restrict__ dst, int n) {
    int i = blockIdx.x * blockDim.x + threadIdx.x;
    if (i < n) dst[i] = (_Float16)src[i];
}

// W[Ksrc][Nsrc] (row-major f32)  ->  Wt[Np][Kp] (f16, zero-padded)
__global__ void prep_wt(const float* __restrict__ W, _Float16* __restrict__ Wt,
                        int Ksrc, int Nsrc, int Kp, int Np) {
    int i = blockIdx.x * blockDim.x + threadIdx.x;
    if (i >= Kp * Np) return;
    int n = i / Kp;
    int k = i - n * Kp;
    float v = (n < Nsrc && k < Ksrc) ? W[(size_t)k * Nsrc + n] : 0.0f;
    Wt[i] = (_Float16)v;
}

// ------------------- fused GEMM + bias + leaky-ReLU ------------------------
// A[M][K] f16 row-major, Wt[Np][K] f16 (N-major), out[M][Np] f16.
// Block = 256 threads = 8 waves. Block tile = 256 rows x 64 cols,
// wave tile = 32 rows x 64 cols (two 16-row groups, 8 WMMAs per K-step).
// B tile (64 cols x 32 K = 4KB) is double-buffered in LDS: one cooperative
// 16B global load per thread per K-step, fragments read via ds_load_b128.

__global__ __launch_bounds__(256)
void gemm_bias_lrelu(const _Float16* __restrict__ A,
                     const _Float16* __restrict__ Wt,
                     const float* __restrict__ bias,
                     _Float16* __restrict__ out,
                     int K, int Np, int Nvalid) {
    __shared__ __align__(32) _Float16 Bs[2][64 * 32];   // 2 x 4KB

    const int lane = threadIdx.x & 31;
    const int wv   = threadIdx.x >> 5;
    const int lo   = lane & 15;
    const int hi   = lane >> 4;
    const int row0 = blockIdx.y * 256 + wv * 32;        // wave owns 32 rows
    const int col0 = blockIdx.x * 64;

    // cooperative B-tile loader mapping: thread t -> (col t>>2, k (t&3)*8)
    const int bcol = threadIdx.x >> 2;
    const int bkof = (threadIdx.x & 3) * 8;
    const _Float16* gB = Wt + (size_t)(col0 + bcol) * K + bkof;
    _Float16* lB0 = &Bs[0][bcol * 32 + bkof];
    _Float16* lB1 = &Bs[1][bcol * 32 + bkof];

    const _Float16* pArow0 = A + (size_t)(row0 + lo) * K;        // rows  0..15
    const _Float16* pArow1 = A + (size_t)(row0 + 16 + lo) * K;   // rows 16..31

    v8f zero;
    #pragma unroll
    for (int e = 0; e < 8; ++e) zero[e] = 0.0f;
    v8f acc[2][4];
    #pragma unroll
    for (int ra = 0; ra < 2; ++ra)
        #pragma unroll
        for (int c = 0; c < 4; ++c) acc[ra][c] = zero;

    // preload first B tile into LDS buffer 0
    *(v8h*)lB0 = *(const v8h*)gB;
    __syncthreads();

    for (int k0 = 0; k0 < K; k0 += 32) {
        const int cur = (k0 >> 5) & 1;
        const bool have_next = (k0 + 32) < K;

        // kick off next B tile's global load early (lands in LDS at loop end)
        v8h bnext;
        if (have_next) {
            bnext = *(const v8h*)(gB + k0 + 32);
            __builtin_prefetch(pArow0 + k0 + 32 + hi * 8, 0, 1);  // A stream hint
        }

        // A fragments (K halves hi*16..hi*16+15 contiguous per lane, matching B)
        AFrag a0, a1;
        a0.h[0] = *(const v8h*)(pArow0 + k0 + hi * 16);
        a0.h[1] = *(const v8h*)(pArow0 + k0 + hi * 16 + 8);
        a1.h[0] = *(const v8h*)(pArow1 + k0 + hi * 16);
        a1.h[1] = *(const v8h*)(pArow1 + k0 + hi * 16 + 8);

        #pragma unroll
        for (int c = 0; c < 4; ++c) {
            // B fragment: col (c*16+lo), K halves hi*16..hi*16+15 -> 32B from LDS
            v16h b = *(const v16h*)&Bs[cur][(c * 16 + lo) * 32 + hi * 16];
            acc[0][c] = wmma_f16(a0.v, b, acc[0][c]);
            acc[1][c] = wmma_f16(a1.v, b, acc[1][c]);
        }

        if (have_next)
            *(v8h*)(cur ? lB0 : lB1) = bnext;   // fill the other buffer
        __syncthreads();
    }

    #pragma unroll
    for (int ra = 0; ra < 2; ++ra) {
        #pragma unroll
        for (int c = 0; c < 4; ++c) {
            int col = col0 + c * 16 + lo;
            bool valid = col < Nvalid;
            float bb = valid ? bias[col] : 0.0f;
            #pragma unroll
            for (int r = 0; r < 8; ++r) {
                float v = acc[ra][c][r] + bb;
                v = (v > 0.0f) ? v : 0.01f * v;
                if (!valid) v = 0.0f;  // keep padded columns exactly zero
                out[(size_t)(row0 + ra * 16 + 8 * hi + r) * Np + col] = (_Float16)v;
            }
        }
    }
}

// ------------------- GEMM3 fused with per-row argmax -----------------------
// A = h2[M][768], Wt = W3t[512][768], bias b3[501]. Each block = 128 rows,
// each wave = 16 rows over all 8 column tiles; wave32 shfl_xor reduction.
// W3t is 0.8MB -> L2/WGP$-resident, so direct global B loads are fine here.

__global__ __launch_bounds__(256)
void gemm_argmax(const _Float16* __restrict__ A,
                 const _Float16* __restrict__ Wt,
                 const float* __restrict__ bias,
                 int* __restrict__ code) {
    const int K = K3;
    const int lane = threadIdx.x & 31;
    const int wv   = threadIdx.x >> 5;
    const int lo   = lane & 15;
    const int hi   = lane >> 4;
    const int row0 = blockIdx.x * 128 + wv * 16;

    const _Float16* pArow = A + (size_t)(row0 + lo) * K;

    float bestv[8];
    int   bestn[8];
    #pragma unroll
    for (int r = 0; r < 8; ++r) { bestv[r] = -3.0e38f; bestn[r] = 0; }

    v8f zero;
    #pragma unroll
    for (int e = 0; e < 8; ++e) zero[e] = 0.0f;

    for (int nt = 0; nt < 8; ++nt) {         // 8 tiles of 64 cols -> 512
        v8f acc[4];
        #pragma unroll
        for (int c = 0; c < 4; ++c) acc[c] = zero;

        for (int k0 = 0; k0 < K; k0 += 32) {
            AFrag a;
            a.h[0] = *(const v8h*)(pArow + k0 + hi * 16);
            a.h[1] = *(const v8h*)(pArow + k0 + hi * 16 + 8);
            #pragma unroll
            for (int c = 0; c < 4; ++c) {
                int col = nt * 64 + c * 16 + lo;
                v16h b = *(const v16h*)(Wt + (size_t)col * K + k0 + hi * 16);
                acc[c] = wmma_f16(a.v, b, acc[c]);
            }
        }

        #pragma unroll
        for (int c = 0; c < 4; ++c) {
            int col = nt * 64 + c * 16 + lo;
            bool valid = col < N3;
            float bb = valid ? bias[col] : 0.0f;
            #pragma unroll
            for (int r = 0; r < 8; ++r) {
                float v = valid ? (acc[c][r] + bb) : -3.0e38f;
                if (v > bestv[r]) { bestv[r] = v; bestn[r] = col; }  // cols ascend: first-max kept
            }
        }
    }

    // reduce across the 16 lanes holding different columns of the same rows
    #pragma unroll
    for (int m = 8; m >= 1; m >>= 1) {
        #pragma unroll
        for (int r = 0; r < 8; ++r) {
            float ov = __shfl_xor(bestv[r], m, 32);
            int   on = __shfl_xor(bestn[r], m, 32);
            if (ov > bestv[r] || (ov == bestv[r] && on < bestn[r])) {
                bestv[r] = ov; bestn[r] = on;
            }
        }
    }

    if (lo == 0) {
        #pragma unroll
        for (int r = 0; r < 8; ++r)
            code[row0 + 8 * hi + r] = bestn[r];
    }
}

// ------------------- bounded cyclic forward fill ---------------------------
// Equivalent to 10000 iterations of code[code==STOP] = roll(code,1)[code==STOP]:
// each stop token takes the nearest non-stop predecessor within distance 10000.

__global__ void forward_fill(const int* __restrict__ code,
                             int* __restrict__ out, int n) {
    int i = blockIdx.x * blockDim.x + threadIdx.x;
    if (i >= n) return;
    int v = code[i];
    if (v == STOP_TOK) {
        int j = i;
        for (int it = 0; it < FILL_ITERS; ++it) {
            j = (j == 0) ? (n - 1) : (j - 1);
            int u = code[j];
            if (u != STOP_TOK) { v = u; break; }
        }
    }
    out[i] = v;
}

// ---------------------------------------------------------------------------

extern "C" void kernel_launch(void* const* d_in, const int* in_sizes, int n_in,
                              void* d_out, int out_size, void* d_ws, size_t ws_size,
                              hipStream_t stream) {
    const float* x  = (const float*)d_in[0];
    const float* W1 = (const float*)d_in[1];
    const float* b1 = (const float*)d_in[2];
    const float* W2 = (const float*)d_in[3];
    const float* b2 = (const float*)d_in[4];
    const float* W3 = (const float*)d_in[5];
    const float* b3 = (const float*)d_in[6];

    char* ws = (char*)d_ws;
    _Float16* xh  = (_Float16*)(ws + OFF_XH);
    _Float16* w1t = (_Float16*)(ws + OFF_W1T);
    _Float16* w2t = (_Float16*)(ws + OFF_W2T);
    _Float16* w3t = (_Float16*)(ws + OFF_W3T);
    _Float16* h1  = (_Float16*)(ws + OFF_H1);
    _Float16* h2  = (_Float16*)(ws + OFF_H2);
    int*      cd  = (int*)(ws + OFF_CODE);
    int*      out = (int*)d_out;

    // 1) convert x to f16 (one pass; f16 x fits in L2 for tile re-reads)
    {
        int n = M_ROWS * K1;
        cvt_f32_to_f16<<<(n + 255) / 256, 256, 0, stream>>>(x, xh, n);
    }
    // 2) transpose+convert weights into [Npad][Kpad] f16 with zero padding
    {
        int n = N1P * K1;
        prep_wt<<<(n + 255) / 256, 256, 0, stream>>>(W1, w1t, K1, N1, K1, N1P);
        n = N2P * K2;
        prep_wt<<<(n + 255) / 256, 256, 0, stream>>>(W2, w2t, N1, N2, K2, N2P);
        n = N3P * K3;
        prep_wt<<<(n + 255) / 256, 256, 0, stream>>>(W3, w3t, N2, N3, K3, N3P);
    }
    // 3) layer 1: h1 = lrelu(x @ W1 + b1)   [32000 x 896]
    {
        dim3 grid(N1P / 64, M_ROWS / 256);
        gemm_bias_lrelu<<<grid, 256, 0, stream>>>(xh, w1t, b1, h1, K1, N1P, N1);
    }
    // 4) layer 2: h2 = lrelu(h1 @ W2 + b2)  [32000 x 768]
    {
        dim3 grid(N2P / 64, M_ROWS / 256);
        gemm_bias_lrelu<<<grid, 256, 0, stream>>>(h1, w2t, b2, h2, K2, N2P, N2);
    }
    // 5) layer 3 + argmax (log_softmax is monotonic -> argmax of logits)
    {
        gemm_argmax<<<M_ROWS / 128, 256, 0, stream>>>(h2, w3t, b3, cd);
    }
    // 6) bounded cyclic forward fill of stop tokens
    {
        forward_fill<<<(M_ROWS + 255) / 256, 256, 0, stream>>>(cd, out, M_ROWS);
    }
}
